// CMSA_12953621365324
// MI455X (gfx1250) — matrix operand order repaired and checked by reading, not compiled, not run on hardware
//
#include <hip/hip_runtime.h>

#define NPIX 4096   // H*W
#define CCH  256    // C
#define RDIM 32     // reduced dim
#define BB   4      // batch

typedef __attribute__((ext_vector_type(16))) __bf16 v16bf;
typedef __attribute__((ext_vector_type(8)))  __bf16 v8bf;
typedef __attribute__((ext_vector_type(8)))  float  v8f;

static __device__ __forceinline__ v8f zero8() {
  v8f z;
#pragma unroll
  for (int i = 0; i < 8; ++i) z[i] = 0.0f;
  return z;
}

static __device__ __forceinline__ v16bf cat8(v8bf lo, v8bf hi) {
  v16bf r;
#pragma unroll
  for (int i = 0; i < 8; ++i) { r[i] = lo[i]; r[i + 8] = hi[i]; }
  return r;
}

static __device__ __forceinline__ v8f wmma_bf16(v16bf a, v16bf b, v8f c) {
  // D = A(16x32 bf16) * B(32x16 bf16) + C(16x16 f32)
  return __builtin_amdgcn_wmma_f32_16x16x32_bf16(false, a, false, b,
                                                 (short)0, c, false, false);
}

// Async copy of one 16B piece per lane into LDS (tracked by ASYNCcnt).
static __device__ __forceinline__ void async_copy16(unsigned int lds_off,
                                                    const void* gsrc) {
  asm volatile("global_load_async_to_lds_b128 %0, %1, off"
               :: "v"(lds_off), "v"((unsigned long long)(uintptr_t)gsrc)
               : "memory");
}

static __device__ __forceinline__ void wait_async0() {
  asm volatile("s_wait_asynccnt 0x0" ::: "memory");
}

// -----------------------------------------------------------------------------
// Kernel 1: projections.  Per (b, feature): [Wq(32);Wk(32);Wv(256)] @ X[256,N].
// Stores Q^T, K^T as [N, 32] bf16 (contiguous rows for WMMA tile loads) and
// V as [C, N] bf16.  Grid: (b*2+f)*256 + ntile blocks; 5 waves each cover the
// 20 output row-tiles (4 per wave).
// -----------------------------------------------------------------------------
__global__ __launch_bounds__(160) void cmsa_proj_kernel(
    const float* __restrict__ xf, const float* __restrict__ yf,
    const float* __restrict__ wqx, const float* __restrict__ bqx,
    const float* __restrict__ wkx, const float* __restrict__ bkx,
    const float* __restrict__ wvx, const float* __restrict__ bvx,
    const float* __restrict__ wqy, const float* __restrict__ bqy,
    const float* __restrict__ wky, const float* __restrict__ bky,
    const float* __restrict__ wvy, const float* __restrict__ bvy,
    __bf16* __restrict__ qtx, __bf16* __restrict__ ktx,
    __bf16* __restrict__ qty, __bf16* __restrict__ kty,
    __bf16* __restrict__ vbx, __bf16* __restrict__ vby)
{
  const int lane  = threadIdx.x & 31;
  const int wave  = threadIdx.x >> 5;      // 0..4
  const int blk   = blockIdx.x;            // (b*2+f)*256 + ntile
  const int ntile = blk & 255;
  const int f     = (blk >> 8) & 1;
  const int b     = blk >> 9;
  const int n0    = ntile * 16;

  const float* X  = (f ? yf : xf) + (size_t)b * CCH * NPIX;
  const float* wq = f ? wqy : wqx; const float* bq = f ? bqy : bqx;
  const float* wk = f ? wky : wkx; const float* bk = f ? bky : bkx;
  const float* wv = f ? wvy : wvx; const float* bv = f ? bvy : bvx;
  __bf16* Qt = (f ? qty : qtx) + (size_t)b * NPIX * RDIM;
  __bf16* Kt = (f ? kty : ktx) + (size_t)b * NPIX * RDIM;
  __bf16* Vb = (f ? vby : vbx) + (size_t)b * CCH * NPIX;

  const int  col   = lane & 15;
  const bool hi    = lane >= 16;
  const int  base8 = hi ? 8 : 0;   // D-layout row base per lane half
  const int  kb16  = hi ? 16 : 0;  // B-layout K base per lane half

  v8f acc[4];
#pragma unroll
  for (int t = 0; t < 4; ++t) acc[t] = zero8();

  for (int c0 = 0; c0 < CCH; c0 += 32) {
    // B tile: X[c0+k, n0+col], k = kb16+e  (B layout: lane=column, K contiguous)
    v16bf Bx;
#pragma unroll
    for (int e = 0; e < 16; ++e)
      Bx[e] = (__bf16)X[(size_t)(c0 + kb16 + e) * NPIX + n0 + col];

#pragma unroll
    for (int t = 0; t < 4; ++t) {
      const int ot = wave * 4 + t;          // output row-tile 0..19
      const float* W; int rowoff;
      if (ot < 2)      { W = wq; rowoff = ot * 16; }
      else if (ot < 4) { W = wk; rowoff = (ot - 2) * 16; }
      else             { W = wv; rowoff = (ot - 4) * 16; }
      const float* wrow = W + (size_t)(rowoff + col) * CCH + c0;
      v16bf Aw;
#pragma unroll
      for (int e = 0; e < 8; ++e) {
        Aw[e]     = (__bf16)wrow[kb16 + e];
        Aw[e + 8] = (__bf16)wrow[kb16 + 8 + e];
      }
      acc[t] = wmma_bf16(Aw, Bx, acc[t]);
    }
  }

  // D layout: lane holds column n=col, rows m = base8 + r (r=0..7)
#pragma unroll
  for (int t = 0; t < 4; ++t) {
    const int ot = wave * 4 + t;
    if (ot < 4) {
      const float* bias = (ot < 2) ? bq : bk;
      const int rowoff  = ((ot < 2) ? ot : (ot - 2)) * 16;
      __bf16* dst = ((ot < 2) ? Qt : Kt) +
                    (size_t)(n0 + col) * RDIM + rowoff + base8;
      v8bf pk;
#pragma unroll
      for (int r = 0; r < 8; ++r)
        pk[r] = (__bf16)(acc[t][r] + bias[rowoff + base8 + r]);
      *(v8bf*)dst = pk;
    } else {
      const int rowoff = (ot - 4) * 16;
#pragma unroll
      for (int r = 0; r < 8; ++r) {
        const int c = rowoff + base8 + r;
        Vb[(size_t)c * NPIX + n0 + col] = (__bf16)(acc[t][r] + bv[c]);
      }
    }
  }
}

// -----------------------------------------------------------------------------
// Kernel 2: fused flash attention + residual.
// Block = 8 waves = one (b, dir, 128-query slab).  Wave w computes S/exp for
// query tile w (published to LDS, which also performs the D->B layout
// transpose) and accumulates O for channel slice [32w, 32w+32) across all 8
// query tiles, so every V A-tile load feeds 8 WMMAs.  K chunks are staged to
// LDS with double-buffered global_load_async_to_lds_b128 (ASYNCcnt), one
// chunk ahead of use.
// -----------------------------------------------------------------------------
__global__ __launch_bounds__(256) void cmsa_flash_kernel(
    const float* __restrict__ xf, const float* __restrict__ yf,
    const __bf16* __restrict__ qtx, const __bf16* __restrict__ ktx,
    const __bf16* __restrict__ qty, const __bf16* __restrict__ kty,
    const __bf16* __restrict__ vbx, const __bf16* __restrict__ vby,
    float* __restrict__ out)
{
  __shared__ __bf16 kbuf[2][32][RDIM];   // [buf][j][d] : 2 x 2KB
  __shared__ __bf16 pbuf[8][16][32];     // [qtile][i][j] : 8KB
  __shared__ float  lbuf[8][16];         // row sums

  const int tid  = threadIdx.x;
  const int lane = tid & 31;
  const int wave = tid >> 5;             // 0..7
  const int blk  = blockIdx.x;           // b*64 + dir*32 + slab
  const int slab = blk & 31;
  const int dir  = (blk >> 5) & 1;       // 0: enhance x (att from y), 1: enhance y
  const int b    = blk >> 6;
  const int i0   = slab * 128;           // first of 128 queries for this block

  const int  col   = lane & 15;
  const bool hi    = lane >= 16;
  const int  base8 = hi ? 8 : 0;
  const int  kb16  = hi ? 16 : 0;
  const int  cbase = wave * 32;          // this wave's 32-channel slice

  const __bf16* Q   = (dir ? qtx : qty) + (size_t)b * NPIX * RDIM;
  const __bf16* Kt  = (dir ? ktx : kty) + (size_t)b * NPIX * RDIM;
  const __bf16* V   = (dir ? vby : vbx) + (size_t)b * CCH * NPIX;
  const float*  res = (dir ? yf  : xf)  + (size_t)b * CCH * NPIX;
  float* outp = out + (size_t)(dir * BB + b) * CCH * NPIX;

  const unsigned int kbuf_off = (unsigned int)(uintptr_t)(void*)&kbuf[0][0][0];

  // B operand for this wave's own query tile: Q[d, myi0+col], d = base8+e (e<8), 16+base8+(e-8) (e>=8)
  const int myi0 = i0 + wave * 16;
  const __bf16* qrow = Q + (size_t)(myi0 + col) * RDIM + base8;
  const v16bf Bq = cat8(*(const v8bf*)qrow, *(const v8bf*)(qrow + 16));

  v8f O[8][2];                            // [qtile][ctile]
#pragma unroll
  for (int it = 0; it < 8; ++it)
#pragma unroll
    for (int ct = 0; ct < 2; ++ct) O[it][ct] = zero8();
  float lacc = 0.0f;
  const v8f zc = zero8();

  // Prologue: async-stage K chunk 0 (first 128 threads move 16B each = 2KB)
  if (tid < 128)
    async_copy16(kbuf_off + (unsigned)tid * 16,
                 Kt + (size_t)(tid >> 2) * RDIM + (tid & 3) * 8);

  for (int j0 = 0; j0 < NPIX; j0 += 32) {
    const int p = (j0 >> 5) & 1;
    wait_async0();                 // this wave's pieces of chunk j0 landed
    __syncthreads();               // all pieces landed; prev iter's LDS reads done

    if (tid < 128 && j0 + 32 < NPIX)  // prefetch next chunk into other buffer
      async_copy16(kbuf_off + (unsigned)(p ^ 1) * 2048 + (unsigned)tid * 16,
                   Kt + (size_t)(j0 + 32 + (tid >> 2)) * RDIM + (tid & 3) * 8);

    // S^T tiles for this wave's query tile from LDS-staged K.
    const __bf16* kb = &kbuf[p][0][0];
    const v16bf A0 = cat8(*(const v8bf*)(kb + col * RDIM + base8),
                          *(const v8bf*)(kb + col * RDIM + 16 + base8));
    const v16bf A1 = cat8(*(const v8bf*)(kb + (16 + col) * RDIM + base8),
                          *(const v8bf*)(kb + (16 + col) * RDIM + 16 + base8));
    const v8f t0 = wmma_bf16(A0, Bq, zc);   // T[j, i] = S[i, j0+j]
    const v8f t1 = wmma_bf16(A1, Bq, zc);

    v8bf p0, p1;
#pragma unroll
    for (int r = 0; r < 8; ++r) {
      const float e0 = __expf(t0[r]);
      const float e1 = __expf(t1[r]);
      lacc += e0 + e1;
      p0[r] = (__bf16)e0;
      p1[r] = (__bf16)e1;
    }
    // D layout: lane holds column i=col, rows j = base8+r (t0) / 16+base8+r (t1)
    *(v8bf*)(&pbuf[wave][col][base8])      = p0;
    *(v8bf*)(&pbuf[wave][col][16 + base8]) = p1;
    __syncthreads();               // P for all 8 query tiles published

    // B operands P^T[j, i] for all 8 query tiles (contiguous 32B per lane)
    v16bf Bp[8];
#pragma unroll
    for (int it = 0; it < 8; ++it)
      Bp[it] = cat8(*(const v8bf*)(&pbuf[it][col][kb16]),
                    *(const v8bf*)(&pbuf[it][col][kb16 + 8]));

    // O[c, i] += V[c, j0+j] * P^T : each V A-tile feeds 8 WMMAs
#pragma unroll
    for (int ct = 0; ct < 2; ++ct) {
      const __bf16* vrow = V + (size_t)(cbase + ct * 16 + col) * NPIX + j0;
      const v16bf Av = cat8(*(const v8bf*)(vrow + kb16),
                            *(const v8bf*)(vrow + kb16 + 8));
#pragma unroll
      for (int it = 0; it < 8; ++it)
        O[it][ct] = wmma_bf16(Av, Bp[it], O[it][ct]);
    }
  }

  // Row sums: lanes col and col+16 jointly cover all j for query i=col.
  const float l = lacc + __shfl_xor(lacc, 16, 32);
  if (lane < 16) lbuf[wave][lane] = l;
  __syncthreads();

#pragma unroll
  for (int it = 0; it < 8; ++it) {
    const float inv = 1.0f / lbuf[it][col];
#pragma unroll
    for (int ct = 0; ct < 2; ++ct) {
#pragma unroll
      for (int r = 0; r < 8; ++r) {
        const int    c   = cbase + ct * 16 + base8 + r;
        const size_t idx = (size_t)c * NPIX + i0 + it * 16 + col;
        outp[idx] = O[it][ct][r] * inv + res[idx];
      }
    }
  }
}

// -----------------------------------------------------------------------------
extern "C" void kernel_launch(void* const* d_in, const int* in_sizes, int n_in,
                              void* d_out, int out_size, void* d_ws, size_t ws_size,
                              hipStream_t stream) {
  (void)in_sizes; (void)n_in; (void)out_size; (void)ws_size;
  const float* xf  = (const float*)d_in[0];
  const float* yf  = (const float*)d_in[1];
  const float* wqx = (const float*)d_in[2];
  const float* bqx = (const float*)d_in[3];
  const float* wkx = (const float*)d_in[4];
  const float* bkx = (const float*)d_in[5];
  const float* wvx = (const float*)d_in[6];
  const float* bvx = (const float*)d_in[7];
  const float* wqy = (const float*)d_in[8];
  const float* bqy = (const float*)d_in[9];
  const float* wky = (const float*)d_in[10];
  const float* bky = (const float*)d_in[11];
  const float* wvy = (const float*)d_in[12];
  const float* bvy = (const float*)d_in[13];
  float* out = (float*)d_out;

  // Workspace: Qt_x | Kt_x | Qt_y | Kt_y (1 MB each) | Vb_x | Vb_y (8 MB each)
  char* ws = (char*)d_ws;
  const size_t qk = (size_t)BB * NPIX * RDIM * 2;  // 1 MB
  const size_t vb = (size_t)BB * CCH * NPIX * 2;   // 8 MB
  __bf16* qtx = (__bf16*)(ws);
  __bf16* ktx = (__bf16*)(ws + qk);
  __bf16* qty = (__bf16*)(ws + 2 * qk);
  __bf16* kty = (__bf16*)(ws + 3 * qk);
  __bf16* vbx = (__bf16*)(ws + 4 * qk);
  __bf16* vby = (__bf16*)(ws + 4 * qk + vb);

  cmsa_proj_kernel<<<dim3(BB * 2 * (NPIX / 16)), dim3(160), 0, stream>>>(
      xf, yf, wqx, bqx, wkx, bkx, wvx, bvx, wqy, bqy, wky, bky, wvy, bvy,
      qtx, ktx, qty, kty, vbx, vby);

  cmsa_flash_kernel<<<dim3(BB * 2 * (NPIX / 128)), dim3(256), 0, stream>>>(
      xf, yf, qtx, ktx, qty, kty, vbx, vby, out);
}
